// MultiHeadedAttention_17867063952095
// MI455X (gfx1250) — matrix operand order repaired and checked by reading, not compiled, not run on hardware
//
#include <hip/hip_runtime.h>

typedef __bf16 bf16_t;
typedef __attribute__((ext_vector_type(16))) __bf16 v16bf;
typedef __attribute__((ext_vector_type(8)))  __bf16 v8bf;
typedef __attribute__((ext_vector_type(8)))  float  v8f;

#define BATCH 4
#define SEQ   2048
#define NH    16
#define HD    64
#define EMB   1024
#define M_TOT (BATCH*SEQ)   // 8192
#define N_QKV (3*EMB)       // 3072

// ---------------------------------------------------------------------------
// WMMA helper: D = A(16x32 bf16) x B(32x16 bf16) + C(16x16 f32)
// ---------------------------------------------------------------------------
__device__ __forceinline__ v8f wmma_bf16(v16bf a, v16bf b, v8f c) {
  return __builtin_amdgcn_wmma_f32_16x16x32_bf16(false, a, false, b, (short)0, c,
                                                 false, false);
}

// A-operand loader from a row-major row pointer:
// lanes 0-15 hold K 0-15, lanes 16-31 hold K 16-31 (matches load_b).
__device__ __forceinline__ v16bf load_a(const bf16_t* __restrict__ row, int kBase,
                                        int lane) {
  return *(const v16bf*)(row + kBase + ((lane < 16) ? 0 : 16));
}

// B-operand loader: source must be column-major (row = N column, contiguous K).
// lane L -> column N = L%16, halves = K 0-15 (lanes<16) / 16-31 (lanes>=16).
__device__ __forceinline__ v16bf load_b(const bf16_t* __restrict__ colRow, int kBase,
                                        int lane) {
  return *(const v16bf*)(colRow + kBase + ((lane < 16) ? 0 : 16));
}

// ---------------------------------------------------------------------------
// fp32 -> bf16 conversion kernels
// ---------------------------------------------------------------------------
__global__ void cvt_bf16(const float* __restrict__ src, bf16_t* __restrict__ dst,
                         int n) {
  int i = blockIdx.x * blockDim.x + threadIdx.x;
  int stride = gridDim.x * blockDim.x;
  for (; i < n; i += stride) dst[i] = (bf16_t)src[i];
}

// dst[c*R + r] = src[r*C + c]  (row-major [R,C] -> row-major [C,R])
__global__ void cvt_bf16_t(const float* __restrict__ src, bf16_t* __restrict__ dst,
                           int R, int C) {
  int i = blockIdx.x * blockDim.x + threadIdx.x;
  int n = R * C;
  int stride = gridDim.x * blockDim.x;
  for (; i < n; i += stride) {
    int r = i / C, c = i - r * C;
    dst[(size_t)c * R + r] = (bf16_t)src[i];
  }
}

// ---------------------------------------------------------------------------
// QKV GEMM: X[8192,1024]bf16 x Wt[3072,1024]bf16 (+bias f32)
// Scatter: Q [B,H,S,D], K [B,H,S,D], Vt [B,H,D,S]  (all bf16)
// One wave computes a 16x64 output strip (A reused across 4 N-tiles).
// ---------------------------------------------------------------------------
__global__ void __launch_bounds__(256) qkv_gemm(
    const bf16_t* __restrict__ X, const bf16_t* __restrict__ Wt,
    const float* __restrict__ bias, bf16_t* __restrict__ Q,
    bf16_t* __restrict__ K, bf16_t* __restrict__ Vt) {
  const int wave = (blockIdx.x * blockDim.x + threadIdx.x) >> 5;
  const int lane = threadIdx.x & 31;
  const int nGroups = N_QKV / 64;          // 48
  const int mTile = wave / nGroups;        // 0..511
  const int nGroup = wave - mTile * nGroups;
  const int mBase = mTile * 16;
  const bf16_t* aRow = X + (size_t)(mBase + (lane & 15)) * EMB;

  v8f acc[4] = {};
  for (int k = 0; k < EMB; k += 32) {
    v16bf a = load_a(aRow, k, lane);
#pragma unroll
    for (int t = 0; t < 4; ++t) {
      const bf16_t* bRow = Wt + (size_t)(nGroup * 64 + t * 16 + (lane & 15)) * EMB;
      acc[t] = wmma_bf16(a, load_b(bRow, k, lane), acc[t]);
    }
  }

  const int mOff = (lane >> 4) * 8;
  const int nl = lane & 15;
#pragma unroll
  for (int t = 0; t < 4; ++t) {
    const int col = nGroup * 64 + t * 16 + nl;
    const float bv = bias[col];
#pragma unroll
    for (int r = 0; r < 8; ++r) {
      const int m = mBase + r + mOff;
      const int b = m >> 11;           // / SEQ
      const int s = m & (SEQ - 1);
      const bf16_t o = (bf16_t)(acc[t][r] + bv);
      if (col < EMB) {                 // Q
        const int h = col >> 6, d = col & 63;
        Q[(((size_t)(b * NH + h) * SEQ + s) << 6) + d] = o;
      } else if (col < 2 * EMB) {      // K (kept row-major: B-source for Q.K^T)
        const int c = col - EMB, h = c >> 6, d = c & 63;
        K[(((size_t)(b * NH + h) * SEQ + s) << 6) + d] = o;
      } else {                         // V transposed: B-source for P.V
        const int c = col - 2 * EMB, h = c >> 6, d = c & 63;
        Vt[(((size_t)(b * NH + h) * HD + d) << 11) + s] = o;
      }
    }
  }
}

// ---------------------------------------------------------------------------
// Flash attention: one wave owns a 16-query tile; streams 32 keys / iter.
// scores = Q.K^T / 8, key-mask -> -1e30, online softmax, O += P.V
// ---------------------------------------------------------------------------
__global__ void __launch_bounds__(256) attn_flash(
    const bf16_t* __restrict__ Q, const bf16_t* __restrict__ K,
    const bf16_t* __restrict__ Vt, const int* __restrict__ Mask,
    bf16_t* __restrict__ AO) {
  __shared__ __align__(32) bf16_t pbuf[8][16 * 32];   // 1KB per wave
  const int warp = threadIdx.x >> 5;
  const int lane = threadIdx.x & 31;
  const int wave = blockIdx.x * 8 + warp;
  const int mTile = wave & 127;            // S/16 tiles
  const int bh = wave >> 7;                // b*NH + h
  const int b = bh >> 4, h = bh & 15;
  const int m0 = mTile * 16;
  const int nl = lane & 15;

  const bf16_t* qBase  = Q  + (size_t)bh * SEQ * HD;
  const bf16_t* kBase  = K  + (size_t)bh * SEQ * HD;
  const bf16_t* vtBase = Vt + (size_t)bh * HD * SEQ;
  bf16_t* myP = &pbuf[warp][0];

  // Loop-invariant A operands: Q tile rows, K-dim split 0-31 / 32-63
  const bf16_t* qRow = qBase + (size_t)(m0 + nl) * HD;
  const v16bf aq0 = load_a(qRow, 0, lane);
  const v16bf aq1 = load_a(qRow, 32, lane);

  v8f o0 = {}, o1 = {}, o2 = {}, o3 = {};
  v8f mrun, lrun;
#pragma unroll
  for (int r = 0; r < 8; ++r) { mrun[r] = -1e30f; lrun[r] = 0.f; }

  for (int n0 = 0; n0 < SEQ; n0 += 32) {
    // --- scores: two 16x16 tiles (keys n0..+15, n0+16..+31), K-dim = 64 ---
    const bf16_t* kRow1 = kBase + (size_t)(n0 + nl) * HD;        // tile 1 col rows
    const bf16_t* kRow2 = kBase + (size_t)(n0 + 16 + nl) * HD;   // tile 2 col rows
    v8f s1 = {}, s2 = {};
    s1 = wmma_bf16(aq0, load_b(kRow1, 0, lane), s1);
    s1 = wmma_bf16(aq1, load_b(kRow1, 32, lane), s1);
    s2 = wmma_bf16(aq0, load_b(kRow2, 0, lane), s2);
    s2 = wmma_bf16(aq1, load_b(kRow2, 32, lane), s2);

    if (n0 + 32 < SEQ) {   // pull next key block toward L2/L0
      __builtin_prefetch(kBase + (size_t)(n0 + 32 + nl) * HD, 0, 1);
      __builtin_prefetch(vtBase + (size_t)nl * SEQ + n0 + 32, 0, 1);
    }

    // --- scale + key mask (column = lane%16) ---
    const int mk1 = Mask[b * SEQ + n0 + nl];
    const int mk2 = Mask[b * SEQ + n0 + 16 + nl];
#pragma unroll
    for (int r = 0; r < 8; ++r) {
      s1[r] = mk1 ? s1[r] * 0.125f : -1e30f;
      s2[r] = mk2 ? s2[r] * 0.125f : -1e30f;
    }

    // --- online softmax: row stats live per (VGPR r, lane-half) slot ---
    v8f tmax;
#pragma unroll
    for (int r = 0; r < 8; ++r) tmax[r] = fmaxf(s1[r], s2[r]);
#pragma unroll
    for (int off = 1; off < 16; off <<= 1)
#pragma unroll
      for (int r = 0; r < 8; ++r)
        tmax[r] = fmaxf(tmax[r], __shfl_xor(tmax[r], off, 16));

    v8f alpha;
#pragma unroll
    for (int r = 0; r < 8; ++r) {
      const float mn = fmaxf(mrun[r], tmax[r]);
      alpha[r] = __expf(mrun[r] - mn);
      s1[r] = __expf(s1[r] - mn);
      s2[r] = __expf(s2[r] - mn);
      mrun[r] = mn;
    }
    v8f rs;
#pragma unroll
    for (int r = 0; r < 8; ++r) rs[r] = s1[r] + s2[r];
#pragma unroll
    for (int off = 1; off < 16; off <<= 1)
#pragma unroll
      for (int r = 0; r < 8; ++r) rs[r] += __shfl_xor(rs[r], off, 16);
#pragma unroll
    for (int r = 0; r < 8; ++r) {
      lrun[r] = lrun[r] * alpha[r] + rs[r];
      o0[r] *= alpha[r]; o1[r] *= alpha[r]; o2[r] *= alpha[r]; o3[r] *= alpha[r];
    }

    // --- C-layout P -> A-layout via LDS (16 rows x 32 keys, bf16) ---
    const int mOff = (lane >> 4) * 8;
#pragma unroll
    for (int r = 0; r < 8; ++r) {
      myP[(r + mOff) * 32 + nl]      = (bf16_t)s1[r];
      myP[(r + mOff) * 32 + 16 + nl] = (bf16_t)s2[r];
    }
    asm volatile("s_wait_dscnt 0" ::: "memory");
    const v16bf ap = load_a(myP + nl * 32, 0, lane);

    // --- O += P.V : 4 d-tiles, B-source = Vt rows (column-major V) ---
    o0 = wmma_bf16(ap, load_b(vtBase + (size_t)(0  + nl) * SEQ + n0, 0, lane), o0);
    o1 = wmma_bf16(ap, load_b(vtBase + (size_t)(16 + nl) * SEQ + n0, 0, lane), o1);
    o2 = wmma_bf16(ap, load_b(vtBase + (size_t)(32 + nl) * SEQ + n0, 0, lane), o2);
    o3 = wmma_bf16(ap, load_b(vtBase + (size_t)(48 + nl) * SEQ + n0, 0, lane), o3);
  }

  // --- finalize: O /= l, store merged-head [B,S,E] bf16 ---
  const int mOff = (lane >> 4) * 8;
  bf16_t* aoBase = AO + ((size_t)b * SEQ + m0) * EMB + h * HD;
#pragma unroll
  for (int r = 0; r < 8; ++r) {
    const float inv = 1.0f / lrun[r];
    const size_t ro = (size_t)(r + mOff) * EMB;
    aoBase[ro + 0  + nl] = (bf16_t)(o0[r] * inv);
    aoBase[ro + 16 + nl] = (bf16_t)(o1[r] * inv);
    aoBase[ro + 32 + nl] = (bf16_t)(o2[r] * inv);
    aoBase[ro + 48 + nl] = (bf16_t)(o3[r] * inv);
  }
}

// ---------------------------------------------------------------------------
// Output projection: AO[8192,1024]bf16 x Wot[1024,1024]bf16 + bias -> f32 out
// ---------------------------------------------------------------------------
__global__ void __launch_bounds__(256) out_proj(
    const bf16_t* __restrict__ AO, const bf16_t* __restrict__ Wot,
    const float* __restrict__ bias, float* __restrict__ out) {
  const int wave = (blockIdx.x * blockDim.x + threadIdx.x) >> 5;
  const int lane = threadIdx.x & 31;
  const int nGroups = EMB / 64;            // 16
  const int mTile = wave / nGroups;
  const int nGroup = wave - mTile * nGroups;
  const int mBase = mTile * 16;
  const bf16_t* aRow = AO + (size_t)(mBase + (lane & 15)) * EMB;

  v8f acc[4] = {};
  for (int k = 0; k < EMB; k += 32) {
    v16bf a = load_a(aRow, k, lane);
#pragma unroll
    for (int t = 0; t < 4; ++t) {
      const bf16_t* bRow = Wot + (size_t)(nGroup * 64 + t * 16 + (lane & 15)) * EMB;
      acc[t] = wmma_bf16(a, load_b(bRow, k, lane), acc[t]);
    }
  }

  const int mOff = (lane >> 4) * 8;
  const int nl = lane & 15;
#pragma unroll
  for (int t = 0; t < 4; ++t) {
    const int col = nGroup * 64 + t * 16 + nl;
    const float bv = bias[col];
#pragma unroll
    for (int r = 0; r < 8; ++r)
      out[(size_t)(mBase + r + mOff) * EMB + col] = acc[t][r] + bv;
  }
}

// ---------------------------------------------------------------------------
extern "C" void kernel_launch(void* const* d_in, const int* in_sizes, int n_in,
                              void* d_out, int out_size, void* d_ws, size_t ws_size,
                              hipStream_t stream) {
  const float* hidden = (const float*)d_in[0];   // [B,S,E] f32
  const float* qkv_w  = (const float*)d_in[1];   // [E,3E]  f32
  const float* qkv_b  = (const float*)d_in[2];   // [3E]    f32
  const float* wo_w   = (const float*)d_in[3];   // [E,E]   f32
  const float* wo_b   = (const float*)d_in[4];   // [E]     f32
  const int*   mask   = (const int*)d_in[5];     // [B,S]   i32
  float* out = (float*)d_out;

  char* ws = (char*)d_ws;
  size_t off = 0;
  auto alloc = [&](size_t bytes) { char* p = ws + off; off += bytes; return p; };
  bf16_t* Xb   = (bf16_t*)alloc((size_t)M_TOT * EMB * 2);       // 16 MB
  bf16_t* Wqkt = (bf16_t*)alloc((size_t)N_QKV * EMB * 2);       //  6 MB [3072,1024]
  bf16_t* Wot  = (bf16_t*)alloc((size_t)EMB * EMB * 2);         //  2 MB [1024,1024]^T
  bf16_t* Qd   = (bf16_t*)alloc((size_t)M_TOT * EMB * 2);       // 16 MB [B,H,S,D]
  bf16_t* Kd   = (bf16_t*)alloc((size_t)M_TOT * EMB * 2);       // 16 MB [B,H,S,D]
  bf16_t* Vt   = (bf16_t*)alloc((size_t)M_TOT * EMB * 2);       // 16 MB [B,H,D,S]
  bf16_t* AO   = (bf16_t*)alloc((size_t)M_TOT * EMB * 2);       // 16 MB [B,S,E]
  (void)ws_size; (void)in_sizes; (void)n_in; (void)out_size;

  cvt_bf16<<<4096, 256, 0, stream>>>(hidden, Xb, M_TOT * EMB);
  cvt_bf16_t<<<4096, 256, 0, stream>>>(qkv_w, Wqkt, EMB, N_QKV);
  cvt_bf16_t<<<2048, 256, 0, stream>>>(wo_w, Wot, EMB, EMB);

  // 512 mTiles x 48 nGroups = 24576 waves, 8 waves/block
  qkv_gemm<<<3072, 256, 0, stream>>>(Xb, Wqkt, qkv_b, Qd, Kd, Vt);
  // 4*16*128 = 8192 waves
  attn_flash<<<1024, 256, 0, stream>>>(Qd, Kd, Vt, mask, AO);
  // 512 mTiles x 16 nGroups = 8192 waves
  out_proj<<<1024, 256, 0, stream>>>(AO, Wot, wo_b, out);
}
